// BitLinear_10376640987248
// MI455X (gfx1250) — compile-verified, matrix operand order checked by reading, not run on hardware
//
#include <hip/hip_runtime.h>

// ---------------------------------------------------------------------------
// BitLinear forward for MI455X (gfx1250, wave32, WMMA + async LDS DMA)
//   out[m,n] = sum_k x[m,k] * (q[n,k] * scale[n, k/128])
// q in {-1,0,+1} is exact in bf16; x split into hi+lo bf16 planes; per-128
// group partial sums via v_wmma_f32_16x16x32_bf16, f32 scale applied by VALU
// FMA at group boundaries. Double-buffered LDS filled with
// global_load_async_to_lds_b128 (ASYNCcnt) when the toolchain provides it.
// ---------------------------------------------------------------------------

typedef __attribute__((ext_vector_type(16))) __bf16 v16bf;
typedef __attribute__((ext_vector_type(8)))  float  v8f;
typedef int b128_t __attribute__((__vector_size__(4 * sizeof(int))));

#define M_DIM 4096   // 2 * 2048
#define N_DIM 4096
#define K_DIM 4096
#define GSZ   128
#define NGRP  (K_DIM / GSZ)   // 32

#define BM 128
#define BN 128
#define BK 128                // == GSZ so one K-tile == one quant group
#define NKB (K_DIM / BK)      // 32

#if __has_builtin(__builtin_amdgcn_global_load_async_to_lds_b128) && \
    __has_builtin(__builtin_amdgcn_s_wait_asynccnt)
#define USE_ASYNC 1
#else
#define USE_ASYNC 0
#endif

#define AS1 __attribute__((address_space(1)))
#define AS3 __attribute__((address_space(3)))

union FragU { v16bf v; uint4 u[2]; };

__device__ __forceinline__ unsigned short f2bf(float f) {
    unsigned u = __float_as_uint(f);
    u += 0x7FFFu + ((u >> 16) & 1u);   // round-to-nearest-even
    return (unsigned short)(u >> 16);
}
__device__ __forceinline__ float bf2f(unsigned short h) {
    return __uint_as_float(((unsigned)h) << 16);
}

// ---------------------------------------------------------------------------
// Kernel 1: groupwise ternary quantization of W (one wave per 128-group).
// ---------------------------------------------------------------------------
__global__ __launch_bounds__(256)
void quantize_weight_kernel(const float* __restrict__ W,
                            unsigned short* __restrict__ qw,
                            float* __restrict__ scales) {
    const int gid  = blockIdx.x * 256 + threadIdx.x;
    const int lane = gid & 31;
    const int grp  = gid >> 5;        // 0 .. N*NGRP-1
    const int n    = grp >> 5;        // NGRP == 32 groups per row
    const int g    = grp & 31;

    const size_t base = (size_t)n * K_DIM + (size_t)g * GSZ + (size_t)lane * 4;
    const float4 w = *(const float4*)(W + base);

    float s = fabsf(w.x) + fabsf(w.y) + fabsf(w.z) + fabsf(w.w);
    #pragma unroll
    for (int off = 16; off > 0; off >>= 1) s += __shfl_xor(s, off, 32);

    const float scale = fmaxf(s * (1.0f / GSZ), 1e-8f);
    const float thr   = 0.5f * scale;

    const float e[4] = {w.x, w.y, w.z, w.w};
    union { unsigned short q[4]; uint2 u; } pk;
    #pragma unroll
    for (int j = 0; j < 4; ++j) {
        const float qv = (e[j] > thr) ? 1.0f : ((e[j] < -thr) ? -1.0f : 0.0f);
        pk.q[j] = f2bf(qv);           // exact: {-1,0,+1}
    }
    *(uint2*)(qw + base) = pk.u;
    if (lane == 0) scales[n * NGRP + g] = scale;
}

// ---------------------------------------------------------------------------
// Kernel 2: split x (f32) into hi/lo bf16 planes: x == hi + lo (~17 mant bits)
// ---------------------------------------------------------------------------
__global__ __launch_bounds__(256)
void convert_x_kernel(const float* __restrict__ x,
                      unsigned short* __restrict__ xh,
                      unsigned short* __restrict__ xl) {
    const size_t gid = (size_t)blockIdx.x * 256 + threadIdx.x;
    const float4 v = ((const float4*)x)[gid];
    const float e[4] = {v.x, v.y, v.z, v.w};
    union { unsigned short s[4]; uint2 u; } h, l;
    #pragma unroll
    for (int j = 0; j < 4; ++j) {
        const unsigned short hb = f2bf(e[j]);
        h.s[j] = hb;
        l.s[j] = f2bf(e[j] - bf2f(hb));
    }
    ((uint2*)xh)[gid] = h.u;
    ((uint2*)xl)[gid] = l.u;
}

// ---------------------------------------------------------------------------
// Tile loader: 3 tiles of 128x128 bf16 (32 KB each) = 24 x b128 per thread.
// Async path: LDS written by DMA, tracked on ASYNCcnt.
// ---------------------------------------------------------------------------
__device__ __forceinline__ void load_tiles(const unsigned short* __restrict__ xh,
                                           const unsigned short* __restrict__ xl,
                                           const unsigned short* __restrict__ qw,
                                           unsigned short* sXh,
                                           unsigned short* sXl,
                                           unsigned short* sWq,
                                           int bm, int bn, int kb, int tid) {
    const unsigned short* gXh = xh + (size_t)(bm * BM) * K_DIM + (size_t)kb * BK;
    const unsigned short* gXl = xl + (size_t)(bm * BM) * K_DIM + (size_t)kb * BK;
    const unsigned short* gWq = qw + (size_t)(bn * BN) * K_DIM + (size_t)kb * BK;
    #pragma unroll
    for (int i = 0; i < 8; ++i) {
        const int c  = i * 256 + tid;    // 0..2047 (128 rows x 16 chunks)
        const int r  = c >> 4;
        const int cc = (c & 15) << 3;
        const size_t go = (size_t)r * K_DIM + cc;
        const int    lo = r * BK + cc;
#if USE_ASYNC
        __builtin_amdgcn_global_load_async_to_lds_b128(
            (AS1 b128_t*)&gXh[go], (AS3 b128_t*)&sXh[lo], 0, 0);
        __builtin_amdgcn_global_load_async_to_lds_b128(
            (AS1 b128_t*)&gXl[go], (AS3 b128_t*)&sXl[lo], 0, 0);
        __builtin_amdgcn_global_load_async_to_lds_b128(
            (AS1 b128_t*)&gWq[go], (AS3 b128_t*)&sWq[lo], 0, 0);
#else
        *(uint4*)&sXh[lo] = *(const uint4*)&gXh[go];
        *(uint4*)&sXl[lo] = *(const uint4*)&gXl[go];
        *(uint4*)&sWq[lo] = *(const uint4*)&gWq[go];
        __builtin_prefetch(&gXh[go + BK], 0, 3);
        __builtin_prefetch(&gWq[go + BK], 0, 3);
#endif
    }
}

// ---------------------------------------------------------------------------
// Kernel 3: double-buffered WMMA GEMM.
// Block tile 128x128x128, 256 threads = 8 waves (4 in M x 2 in N).
// Wave patch 32x64 = 8 C tiles; 64 WMMAs per wave per K-block.
// ---------------------------------------------------------------------------
__global__ __launch_bounds__(256)
void bitlinear_gemm_kernel(const unsigned short* __restrict__ xh,
                           const unsigned short* __restrict__ xl,
                           const unsigned short* __restrict__ qw,
                           const float* __restrict__ scales,
                           float* __restrict__ out) {
    __shared__ __align__(16) unsigned short sXh[2][BM * BK];   // 2 x 32 KB
    __shared__ __align__(16) unsigned short sXl[2][BM * BK];   // 2 x 32 KB
    __shared__ __align__(16) unsigned short sWq[2][BN * BK];   // 2 x 32 KB

    const int tid   = threadIdx.x;
    const int lane  = tid & 31;
    const int wv    = tid >> 5;            // 0..7
    const int wm    = wv >> 1;             // 4 waves along M
    const int wn    = wv & 1;              // 2 waves along N
    const int bm    = blockIdx.y;
    const int bn    = blockIdx.x;

    const int lr    = lane & 15;           // row/col within 16
    const int khalf = (lane >> 4) << 3;    // 0 or 8 (16-bit A/B layout K split)

    const v8f vzero = {0.f, 0.f, 0.f, 0.f, 0.f, 0.f, 0.f, 0.f};
    v8f facc[2][4];
    #pragma unroll
    for (int a = 0; a < 2; ++a)
        #pragma unroll
        for (int b = 0; b < 4; ++b) facc[a][b] = vzero;

    // Prologue: stage tile 0 into buffer 0.
    load_tiles(xh, xl, qw, sXh[0], sXl[0], sWq[0], bm, bn, 0, tid);

    for (int kb = 0; kb < NKB; ++kb) {
        const int cur = kb & 1;

        // Stream next tile into the other buffer, then wait for current tile.
        if (kb + 1 < NKB) {
            load_tiles(xh, xl, qw, sXh[cur ^ 1], sXl[cur ^ 1], sWq[cur ^ 1],
                       bm, bn, kb + 1, tid);
#if USE_ASYNC
            __builtin_amdgcn_s_wait_asynccnt(24);   // 24 in flight => tile kb done
#endif
        } else {
#if USE_ASYNC
            __builtin_amdgcn_s_wait_asynccnt(0);
#endif
        }
        __syncthreads();

        v8f gacc[2][4];
        #pragma unroll
        for (int a = 0; a < 2; ++a)
            #pragma unroll
            for (int b = 0; b < 4; ++b) gacc[a][b] = vzero;

        #pragma unroll
        for (int ks = 0; ks < 4; ++ks) {
            const int kofs = ks * 32 + khalf;  // lanes<16: K 0-7/16-23; >=16: 8-15/24-31

            FragU bq[4];
            #pragma unroll
            for (int tn = 0; tn < 4; ++tn) {
                const unsigned short* p =
                    &sWq[cur][(wn * 64 + tn * 16 + lr) * BK + kofs];
                bq[tn].u[0] = *(const uint4*)p;
                bq[tn].u[1] = *(const uint4*)(p + 16);
            }
            FragU ah[2], al[2];
            #pragma unroll
            for (int tm = 0; tm < 2; ++tm) {
                const int row = (wm * 32 + tm * 16 + lr) * BK + kofs;
                ah[tm].u[0] = *(const uint4*)&sXh[cur][row];
                ah[tm].u[1] = *(const uint4*)&sXh[cur][row + 16];
                al[tm].u[0] = *(const uint4*)&sXl[cur][row];
                al[tm].u[1] = *(const uint4*)&sXl[cur][row + 16];
            }

            #pragma unroll
            for (int tm = 0; tm < 2; ++tm)
                #pragma unroll
                for (int tn = 0; tn < 4; ++tn) {
                    gacc[tm][tn] = __builtin_amdgcn_wmma_f32_16x16x32_bf16(
                        false, ah[tm].v, false, bq[tn].v,
                        (short)0, gacc[tm][tn], false, false);
                    gacc[tm][tn] = __builtin_amdgcn_wmma_f32_16x16x32_bf16(
                        false, al[tm].v, false, bq[tn].v,
                        (short)0, gacc[tm][tn], false, false);
                }
        }

        // facc += gacc * scale[n, kb]   (f32, exact group scaling)
        #pragma unroll
        for (int tm = 0; tm < 2; ++tm)
            #pragma unroll
            for (int tn = 0; tn < 4; ++tn) {
                const int ncol = bn * BN + wn * 64 + tn * 16 + lr;
                const float s = scales[(size_t)ncol * NGRP + kb];
                #pragma unroll
                for (int r = 0; r < 8; ++r)
                    facc[tm][tn][r] = fmaf(gacc[tm][tn][r], s, facc[tm][tn][r]);
            }
        __syncthreads();   // all waves done with buffer `cur` before refill
    }

    // Epilogue: C layout — VGPR r: lanes 0-15 hold M=r, lanes 16-31 hold M=r+8
    const int mhi = (lane >> 4) << 3;
    #pragma unroll
    for (int tm = 0; tm < 2; ++tm)
        #pragma unroll
        for (int tn = 0; tn < 4; ++tn) {
            const int ncol = bn * BN + wn * 64 + tn * 16 + lr;
            const int mrow = bm * BM + wm * 32 + tm * 16 + mhi;
            #pragma unroll
            for (int r = 0; r < 8; ++r)
                out[(size_t)(mrow + r) * N_DIM + ncol] = facc[tm][tn][r];
        }
}

// ---------------------------------------------------------------------------
extern "C" void kernel_launch(void* const* d_in, const int* in_sizes, int n_in,
                              void* d_out, int out_size, void* d_ws, size_t ws_size,
                              hipStream_t stream) {
    const float* x = (const float*)d_in[0];     // [2,2048,4096] f32
    const float* w = (const float*)d_in[1];     // [4096,4096]   f32
    float* out = (float*)d_out;                 // [2,2048,4096] f32

    char* ws = (char*)d_ws;
    const size_t qw_bytes = (size_t)N_DIM * K_DIM * 2;
    const size_t xp_bytes = (size_t)M_DIM * K_DIM * 2;
    unsigned short* qw     = (unsigned short*)(ws);
    unsigned short* xh     = (unsigned short*)(ws + qw_bytes);
    unsigned short* xl     = (unsigned short*)(ws + qw_bytes + xp_bytes);
    float*          scales = (float*)        (ws + qw_bytes + 2 * xp_bytes);

    // 1) quantize W: N*NGRP groups, one wave each
    quantize_weight_kernel<<<(N_DIM * NGRP * 32) / 256, 256, 0, stream>>>(w, qw, scales);
    // 2) split x into hi/lo bf16 planes (4 f32 per thread)
    convert_x_kernel<<<(int)(((size_t)M_DIM * K_DIM / 4) / 256), 256, 0, stream>>>(x, xh, xl);
    // 3) double-buffered WMMA GEMM
    dim3 grid(N_DIM / BN, M_DIM / BM);
    bitlinear_gemm_kernel<<<grid, 256, 0, stream>>>(xh, xl, qw, scales, out);
}
